// GraphConv_12824772346521
// MI455X (gfx1250) — matrix operand order repaired and checked by reading, not compiled, not run on hardware
//
#include <hip/hip_runtime.h>
#include <hip/hip_bf16.h>

#define N_NODES 50000
#define E_EDGES 600000
#define D_IN    128
#define D_HID   256
#define D_OUT   128
#define BN_EPS  1e-5f

typedef __attribute__((ext_vector_type(16))) __bf16 v16bf;
typedef __attribute__((ext_vector_type(8)))  __bf16 v8bf;
typedef __attribute__((ext_vector_type(8)))  float  v8f;
typedef __attribute__((ext_vector_type(4)))  int    v4i;

#if defined(__gfx1250__) && __has_builtin(__builtin_amdgcn_global_load_async_to_lds_b128)
#define HAVE_ASYNC_LDS 1
#else
#define HAVE_ASYNC_LDS 0
#endif

typedef __attribute__((address_space(1))) v4i gv4i;   // global (HIP "__device__") AS
typedef __attribute__((address_space(3))) v4i lv4i;   // LDS ("__shared__") AS

// Assemble a 16-element bf16 WMMA fragment from two contiguous 8-elem runs.
static __device__ inline v16bf ld_frag(const __bf16* lo, const __bf16* hi) {
    v8bf a = *(const v8bf*)lo;
    v8bf b = *(const v8bf*)hi;
    return __builtin_shufflevector(a, b, 0,1,2,3,4,5,6,7,8,9,10,11,12,13,14,15);
}

// Convert 8 consecutive f32 (16B-aligned) to 8 bf16 fragment elements.
static __device__ inline v8bf cvt8(const float* p) {
    float4 a = *(const float4*)p;
    float4 b = *(const float4*)(p + 4);
    v8bf r;
    r[0] = (__bf16)a.x; r[1] = (__bf16)a.y; r[2] = (__bf16)a.z; r[3] = (__bf16)a.w;
    r[4] = (__bf16)b.x; r[5] = (__bf16)b.y; r[6] = (__bf16)b.z; r[7] = (__bf16)b.w;
    return r;
}

// Stage `bytes` (multiple of 16*blockDim) from global to LDS, 16B chunks.
static __device__ inline void stage_tile(const char* gsrc, char* lds, int bytes,
                                         int tid, int nthreads) {
#if HAVE_ASYNC_LDS
    for (int off = tid * 16; off < bytes; off += nthreads * 16) {
        __builtin_amdgcn_global_load_async_to_lds_b128(
            (gv4i*)(gsrc + off), (lv4i*)(lds + off), 0, 0);
    }
    __builtin_amdgcn_s_wait_asynccnt(0);
#else
    const uint4* s = (const uint4*)gsrc;
    uint4* d = (uint4*)lds;
    for (int i = tid; i < bytes / 16; i += nthreads) d[i] = s[i];
#endif
    __syncthreads();
}

// ---------------- init: h = x ----------------
__global__ void __launch_bounds__(256) k_copy(const float4* __restrict__ x4,
                                              float4* __restrict__ h4) {
    size_t i = (size_t)blockIdx.x * 256 + threadIdx.x;   // N*128/4 = 1.6M
    h4[i] = x4[i];
}

// -------- weight transpose + bf16 convert, zero BN accumulators --------
__global__ void __launch_bounds__(256) k_prep(const float* __restrict__ W1,
                                              const float* __restrict__ W2,
                                              __bf16* __restrict__ W1T,
                                              __bf16* __restrict__ W2T,
                                              float* __restrict__ bnsum,
                                              float* __restrict__ bnsq) {
    int idx = blockIdx.x * 256 + threadIdx.x;            // 128 blocks -> 32768
    if (idx < D_IN * D_HID) {
        int n = idx >> 7, k = idx & 127;                 // W1T[n*128+k]
        W1T[idx] = (__bf16)W1[k * D_HID + n];
        int n2 = idx >> 8, k2 = idx & 255;               // W2T[n*256+k]
        W2T[idx] = (__bf16)W2[k2 * D_OUT + n2];
    }
    if (idx < D_OUT) { bnsum[idx] = 0.f; bnsq[idx] = 0.f; }
}

// ---------------- edge scatter-add: h[dst] += x[src] ----------------
__global__ void __launch_bounds__(256) k_scatter(const int* __restrict__ src,
                                                 const int* __restrict__ dst,
                                                 const float4* __restrict__ x4,
                                                 float* __restrict__ h) {
    int wave = blockIdx.x * 8 + (threadIdx.x >> 5);      // one wave per edge
    int lane = threadIdx.x & 31;
    int s = src[wave];
    int d = dst[wave];
    float4 v = x4[(size_t)s * 32 + lane];
    float* hp = h + (size_t)d * D_IN + lane * 4;
    atomicAdd(hp + 0, v.x);
    atomicAdd(hp + 1, v.y);
    atomicAdd(hp + 2, v.z);
    atomicAdd(hp + 3, v.w);
}

// -------- GEMM1: hid = relu(h @ W1 + b1), bf16 out; full 256 cols/block ----
__global__ void __launch_bounds__(128) k_gemm1(const float* __restrict__ h,
                                               const __bf16* __restrict__ W1T,
                                               const float* __restrict__ b1,
                                               __bf16* __restrict__ hid) {
    __shared__ float Asf[16 * D_IN];                     // 8 KB raw f32 tile
    int tid = threadIdx.x;
    int r0 = blockIdx.x * 16;                            // 3125 row tiles (exact)

    stage_tile((const char*)(h + (size_t)r0 * D_IN), (char*)Asf,
               16 * D_IN * 4, tid, 128);

    int wv = tid >> 5, lane = tid & 31;
    int half = lane >> 4;
    const float* arow = &Asf[(lane & 15) * D_IN];

    v8f acc[4] = {};
    #pragma unroll
    for (int kk = 0; kk < D_IN / 32; ++kk) {
        int kb = kk * 32;
        v16bf a = __builtin_shufflevector(
            cvt8(&arow[kb + half * 8]), cvt8(&arow[kb + 16 + half * 8]),
            0,1,2,3,4,5,6,7,8,9,10,11,12,13,14,15);
        #pragma unroll
        for (int j = 0; j < 4; ++j) {                    // 4 col tiles per wave
            int n = j * 64 + wv * 16 + (lane & 15);
            const __bf16* wrow = W1T + (size_t)n * D_IN;
            v16bf b = ld_frag(&wrow[kb + half * 8], &wrow[kb + 16 + half * 8]);
            acc[j] = __builtin_amdgcn_wmma_f32_16x16x32_bf16(
                false, a, false, b, (short)0, acc[j], false, false);
        }
    }

    #pragma unroll
    for (int j = 0; j < 4; ++j) {
        int n = j * 64 + wv * 16 + (lane & 15);
        float bias = b1[n];
        #pragma unroll
        for (int r = 0; r < 8; ++r) {
            int m = r + half * 8;
            float v = acc[j][r] + bias;
            v = v > 0.f ? v : 0.f;
            hid[(size_t)(r0 + m) * D_HID + n] = (__bf16)v;
        }
    }
}

// -------- GEMM2: h2 = hid @ W2 + b2, f32 out; full 128 cols/block ----------
__global__ void __launch_bounds__(128) k_gemm2(const __bf16* __restrict__ hid,
                                               const __bf16* __restrict__ W2T,
                                               const float* __restrict__ b2,
                                               float* __restrict__ h2) {
    __shared__ __bf16 As[16 * D_HID];                    // 8 KB bf16 tile
    int tid = threadIdx.x;
    int r0 = blockIdx.x * 16;

    stage_tile((const char*)(hid + (size_t)r0 * D_HID), (char*)As,
               16 * D_HID * 2, tid, 128);

    int wv = tid >> 5, lane = tid & 31;
    int half = lane >> 4;
    const __bf16* arow = &As[(lane & 15) * D_HID];

    v8f acc[2] = {};
    #pragma unroll
    for (int kk = 0; kk < D_HID / 32; ++kk) {
        int kb = kk * 32;
        v16bf a = ld_frag(&arow[kb + half * 8], &arow[kb + 16 + half * 8]);
        #pragma unroll
        for (int j = 0; j < 2; ++j) {                    // 2 col tiles per wave
            int n = j * 64 + wv * 16 + (lane & 15);
            const __bf16* wrow = W2T + (size_t)n * D_HID;
            v16bf b = ld_frag(&wrow[kb + half * 8], &wrow[kb + 16 + half * 8]);
            acc[j] = __builtin_amdgcn_wmma_f32_16x16x32_bf16(
                false, a, false, b, (short)0, acc[j], false, false);
        }
    }

    #pragma unroll
    for (int j = 0; j < 2; ++j) {
        int n = j * 64 + wv * 16 + (lane & 15);
        float bias = b2[n];
        #pragma unroll
        for (int r = 0; r < 8; ++r) {
            int m = r + half * 8;
            h2[(size_t)(r0 + m) * D_OUT + n] = acc[j][r] + bias;
        }
    }
}

// ---------------- BN stats: per-channel sum & sumsq ----------------
__global__ void __launch_bounds__(256) k_bnstats(const float* __restrict__ h2,
                                                 float* __restrict__ bnsum,
                                                 float* __restrict__ bnsq) {
    __shared__ float ssum[256];
    __shared__ float ssq[256];
    int tid = threadIdx.x;
    int c = tid & 127;
    int rh = tid >> 7;                                   // 2 row phases
    int rbase = blockIdx.x * 200;                        // 250 blocks * 200 rows
    float s = 0.f, q = 0.f;
    for (int i = 0; i < 100; ++i) {
        float v = h2[(size_t)(rbase + i * 2 + rh) * D_OUT + c];
        s += v; q += v * v;
    }
    ssum[tid] = s; ssq[tid] = q;
    __syncthreads();
    if (tid < 128) {
        atomicAdd(&bnsum[c], ssum[tid] + ssum[tid + 128]);
        atomicAdd(&bnsq[c],  ssq[tid]  + ssq[tid + 128]);
    }
}

// ---------------- BN apply + ReLU ----------------
__global__ void __launch_bounds__(256) k_apply(const float* __restrict__ h2,
                                               const float* __restrict__ bnsum,
                                               const float* __restrict__ bnsq,
                                               const float* __restrict__ gamma,
                                               const float* __restrict__ beta,
                                               float* __restrict__ out) {
    size_t idx = (size_t)blockIdx.x * 256 + threadIdx.x; // N*128 elements
    int c = (int)(idx & 127);
    const float invN = 1.0f / (float)N_NODES;
    float mean = bnsum[c] * invN;
    float var = bnsq[c] * invN - mean * mean;
    float scale = gamma[c] * rsqrtf(var + BN_EPS);
    float v = (h2[idx] - mean) * scale + beta[c];
    out[idx] = v > 0.f ? v : 0.f;
}

extern "C" void kernel_launch(void* const* d_in, const int* in_sizes, int n_in,
                              void* d_out, int out_size, void* d_ws, size_t ws_size,
                              hipStream_t stream) {
    const float* x     = (const float*)d_in[0];
    const int*   eidx  = (const int*)d_in[1];   // [2][E], int32
    const float* W1    = (const float*)d_in[2];
    const float* b1    = (const float*)d_in[3];
    const float* W2    = (const float*)d_in[4];
    const float* b2    = (const float*)d_in[5];
    const float* gamma = (const float*)d_in[6];
    const float* beta  = (const float*)d_in[7];
    float* out = (float*)d_out;

    // Workspace layout
    char* ws = (char*)d_ws;
    float*  h    = (float*)ws;                                   ws += (size_t)N_NODES * D_IN * 4;   // 25.6 MB
    __bf16* hid  = (__bf16*)ws;                                  ws += (size_t)N_NODES * D_HID * 2;  // 25.6 MB
    float*  h2   = (float*)ws;                                   ws += (size_t)N_NODES * D_OUT * 4;  // 25.6 MB
    __bf16* W1T  = (__bf16*)ws;                                  ws += (size_t)D_IN * D_HID * 2;
    __bf16* W2T  = (__bf16*)ws;                                  ws += (size_t)D_HID * D_OUT * 2;
    float*  bnsum = (float*)ws;                                  ws += D_OUT * 4;
    float*  bnsq  = (float*)ws;

    const int* src = eidx;
    const int* dst = eidx + E_EDGES;

    // 1. h = x
    k_copy<<<(N_NODES * D_IN) / 4 / 256, 256, 0, stream>>>((const float4*)x, (float4*)h);
    // 2. weights -> bf16 transposed; BN accumulators = 0 (every launch)
    k_prep<<<(D_IN * D_HID + 255) / 256, 256, 0, stream>>>(W1, W2, W1T, W2T, bnsum, bnsq);
    // 3. scatter-add over edges (one wave per edge)
    k_scatter<<<E_EDGES / 8, 256, 0, stream>>>(src, dst, (const float4*)x, h);
    // 4. GEMM1 + bias + ReLU -> hid (bf16); one block covers all 256 cols
    k_gemm1<<<N_NODES / 16, 128, 0, stream>>>(h, W1T, b1, hid);
    // 5. GEMM2 + bias -> h2 (f32); one block covers all 128 cols
    k_gemm2<<<N_NODES / 16, 128, 0, stream>>>(hid, W2T, b2, h2);
    // 6. BN stats
    k_bnstats<<<N_NODES / 200, 256, 0, stream>>>(h2, bnsum, bnsq);
    // 7. BN apply + ReLU
    k_apply<<<(N_NODES * D_OUT) / 256, 256, 0, stream>>>(h2, bnsum, bnsq, gamma, beta, out);
}
